// TransformerLayer_32684701122642
// MI455X (gfx1250) — matrix operand order repaired and checked by reading, not compiled, not run on hardware
//
#include <hip/hip_runtime.h>
#include <math.h>

// ---------------------------------------------------------------------------
// Types for CDNA5 WMMA (gfx1250, wave32)
// ---------------------------------------------------------------------------
typedef __attribute__((ext_vector_type(16))) __bf16 v16bf;
typedef __attribute__((ext_vector_type(8)))  float  v8f;

union FragB { v16bf v; unsigned u[8]; };
union FragC { v8f   v; float    f[8]; };

#define BB 2
#define SS 2048
#define DD 1024
#define HH 16
#define HD 64
#define FFD 4096
#define NTOK (BB*SS)   // 4096 tokens

__device__ __forceinline__ unsigned short f2bfu(float f) {
  unsigned u = __float_as_uint(f);
  u = (u + 0x7fffu + ((u >> 16) & 1u)) >> 16;   // round-to-nearest-even
  return (unsigned short)u;
}

__device__ __forceinline__ v8f wmma32(v16bf a, v16bf b, v8f c) {
  return __builtin_amdgcn_wmma_f32_16x16x32_bf16(false, a, false, b,
                                                 (short)0, c, false, false);
}

// A-operand fragment (16x32 bf16): caller passes this lane's row pointer at
// the k-origin.  half==0: u[0..3]=k0..7, u[4..7]=k16..23; half==1: k8..15,k24..31
__device__ __forceinline__ v16bf load_fragA(const unsigned short* rowp) {
  const int half = (threadIdx.x >> 4) & 1;
  const unsigned short* p0 = rowp + half * 8;
  const unsigned short* p1 = rowp + 16 + half * 8;
  FragB f;
#pragma unroll
  for (int j = 0; j < 4; ++j) {
    f.u[j]     = *(const unsigned*)(p0 + 2 * j);
    f.u[4 + j] = *(const unsigned*)(p1 + 2 * j);
  }
  return f.v;
}

// B-operand fragment (32x16 bf16) from LDS stored as [col][k], row stride
// `stride`; lane c holds column (lane&15), half==0: k0..7,k16..23; half==1: k8..15,k24..31.
__device__ __forceinline__ v16bf load_fragB(const unsigned short* base, int stride) {
  const int lane = threadIdx.x & 31;
  const unsigned short* p = base + (lane & 15) * stride + (lane >> 4) * 8;
  FragB f;
#pragma unroll
  for (int j = 0; j < 4; ++j) {
    f.u[j]     = *(const unsigned*)(p + 2 * j);
    f.u[4 + j] = *(const unsigned*)(p + 16 + 2 * j);
  }
  return f.v;
}

// ---------------------------------------------------------------------------
// CDNA5 async global->LDS copy (16 bytes per lane), ASYNCcnt-tracked
// ---------------------------------------------------------------------------
__device__ __forceinline__ void async_ld16(unsigned short* ldsp,
                                           const unsigned short* g) {
  unsigned loff = (unsigned)(uintptr_t)ldsp;  // low 32 bits of generic ptr = LDS offset
  asm volatile("global_load_async_to_lds_b128 %0, %1, off"
               :: "v"(loff), "v"(g) : "memory");
}

template <int N>
__device__ __forceinline__ void wait_async() {
  asm volatile("s_wait_asynccnt %0" :: "i"(N) : "memory");
}

// ---------------------------------------------------------------------------
// Weight prep: f32 [K,N] -> bf16 transposed [N,K] (LDS-tiled transpose)
// ---------------------------------------------------------------------------
__global__ __launch_bounds__(256) void convert_transpose_bf16(
    const float* __restrict__ in, unsigned short* __restrict__ out,
    int K, int N) {
  __shared__ float t[32][33];
  const int k0 = blockIdx.y * 32, n0 = blockIdx.x * 32;
  const int tx = threadIdx.x & 31, ty = threadIdx.x >> 5;  // ty in 0..7
#pragma unroll
  for (int j = 0; j < 32; j += 8)
    t[ty + j][tx] = in[(size_t)(k0 + ty + j) * N + n0 + tx];
  __syncthreads();
#pragma unroll
  for (int j = 0; j < 32; j += 8)
    out[(size_t)(n0 + ty + j) * K + k0 + tx] = f2bfu(t[tx][ty + j]);
}

// ---------------------------------------------------------------------------
// LayerNorm over D=1024, one row per block (256 threads), bf16 output
// ---------------------------------------------------------------------------
__global__ __launch_bounds__(256) void layernorm_bf16_kernel(
    const float* __restrict__ x, const float* __restrict__ g,
    const float* __restrict__ be, unsigned short* __restrict__ out) {
  const int row = blockIdx.x;
  const int tid = threadIdx.x;
  const float* xr = x + (size_t)row * DD;
  float v[4], s = 0.f, sq = 0.f;
#pragma unroll
  for (int k = 0; k < 4; ++k) {
    v[k] = xr[tid + k * 256];
    s += v[k];
    sq += v[k] * v[k];
  }
#pragma unroll
  for (int off = 16; off; off >>= 1) {
    s  += __shfl_xor(s, off);
    sq += __shfl_xor(sq, off);
  }
  __shared__ float ss[8], sq8[8];
  if ((tid & 31) == 0) { ss[tid >> 5] = s; sq8[tid >> 5] = sq; }
  __syncthreads();
  float ts = 0.f, tq = 0.f;
#pragma unroll
  for (int i = 0; i < 8; ++i) { ts += ss[i]; tq += sq8[i]; }
  const float mu  = ts * (1.0f / DD);
  const float var = tq * (1.0f / DD) - mu * mu;
  const float rs  = rsqrtf(var + 1e-12f);
#pragma unroll
  for (int k = 0; k < 4; ++k) {
    int c = tid + k * 256;
    out[(size_t)row * DD + c] = f2bfu((v[k] - mu) * rs * g[c] + be[c]);
  }
}

// ---------------------------------------------------------------------------
// bf16 WMMA GEMM:  out[M,N] = A[M,K](bf16) @ Wt[N,K](bf16, pre-transposed)
//                  + bias, optional GELU / f32-residual.
// OUTMODE: 0 = f32 row-major, 1 = bf16 row-major, 2 = bf16 [b,h,hd,s] (for V)
// 256 threads (8 waves), BM=128,BN=128,BK=32, double-buffered async LDS.
// ---------------------------------------------------------------------------
template <bool GELU, bool RESID, int OUTMODE>
__global__ __launch_bounds__(256) void gemm_bf16_kernel(
    const unsigned short* __restrict__ A, const unsigned short* __restrict__ Wt,
    const float* __restrict__ bias, const float* __restrict__ resid,
    void* __restrict__ outp, int M, int N, int K) {
  constexpr int BK = 32, AST = BK + 8;  // row stride 40 elems = 80 B (16B-mult)
  __shared__ alignas(16) unsigned short As[2][128 * AST];  // [m][k]
  __shared__ alignas(16) unsigned short Bs[2][128 * AST];  // [n][k]

  const int tid = threadIdx.x;
  const int lane = tid & 31, col16 = lane & 15, half = lane >> 4;
  const int w = tid >> 5, wm = w & 3, wn = w >> 2;
  const int m0 = blockIdx.y * 128, n0 = blockIdx.x * 128;

  FragC acc[2][4];
#pragma unroll
  for (int mt = 0; mt < 2; ++mt)
#pragma unroll
    for (int nt = 0; nt < 4; ++nt)
#pragma unroll
      for (int i = 0; i < 8; ++i) acc[mt][nt].f[i] = 0.f;

  // issue one 128x32 A-tile + 128x32 B-tile: 4 async b128 ops per thread
  auto issue_tile = [&](int buf, int k0) {
#pragma unroll
    for (int it = 0; it < 2; ++it) {
      int c = tid + it * 256;            // 512 chunks of 8 bf16 (16 B)
      int r = c >> 2, c8 = (c & 3) * 8;  // 4 chunks per 32-elem row
      async_ld16(&As[buf][r * AST + c8], A  + (size_t)(m0 + r) * K + k0 + c8);
      async_ld16(&Bs[buf][r * AST + c8], Wt + (size_t)(n0 + r) * K + k0 + c8);
    }
  };

  const int nk = K / BK;
  issue_tile(0, 0);
  for (int kt = 0; kt < nk; ++kt) {
    const int cur = kt & 1;
    if (kt + 1 < nk) { issue_tile(cur ^ 1, (kt + 1) * BK); wait_async<4>(); }
    else             { wait_async<0>(); }
    __syncthreads();   // all waves' DMA for buf[cur] landed

    v16bf af[2];
#pragma unroll
    for (int mt = 0; mt < 2; ++mt)
      af[mt] = load_fragA(&As[cur][(wm * 32 + mt * 16 + col16) * AST]);
    v16bf bfr[4];
#pragma unroll
    for (int nt = 0; nt < 4; ++nt)
      bfr[nt] = load_fragB(&Bs[cur][(wn * 64 + nt * 16) * AST], AST);
#pragma unroll
    for (int mt = 0; mt < 2; ++mt)
#pragma unroll
      for (int nt = 0; nt < 4; ++nt)
        acc[mt][nt].v = wmma32(af[mt], bfr[nt], acc[mt][nt].v);
    __syncthreads();   // reads of buf[cur] done before it is re-issued
  }

  // epilogue: C layout row = i + 8*half, col = lane&15
#pragma unroll
  for (int nt = 0; nt < 4; ++nt) {
    const int gc = n0 + wn * 64 + nt * 16 + col16;
    const float bv = bias[gc];
#pragma unroll
    for (int mt = 0; mt < 2; ++mt) {
#pragma unroll
      for (int i = 0; i < 8; ++i) {
        const int gr = m0 + wm * 32 + mt * 16 + i + 8 * half;
        float val = acc[mt][nt].f[i] + bv;
        if (RESID) val += resid[(size_t)gr * N + gc];
        if (GELU)  val = 0.5f * val * (1.0f + erff(val * 0.70710678118654752f));
        if (OUTMODE == 0) {
          ((float*)outp)[(size_t)gr * N + gc] = val;
        } else if (OUTMODE == 1) {
          ((unsigned short*)outp)[(size_t)gr * N + gc] = f2bfu(val);
        } else {  // [b,h,hd,s] layout for V
          const int hh = gc >> 6, hd = gc & 63;
          const int bb = gr >> 11, sI = gr & 2047;
          ((unsigned short*)outp)[(((size_t)bb * HH + hh) * HD + hd) * SS + sI] =
              f2bfu(val);
        }
      }
    }
  }
}

// ---------------------------------------------------------------------------
// Flash attention.  Q,K in token-major [B*S, D] bf16; V pre-transposed
// [b,h,hd,s] bf16.  256 threads = 8 waves, 128 query rows per block,
// 64-key tiles double-buffered via async global->LDS DMA.
// Online softmax in accumulator layout with xor-16 butterflies.
// ---------------------------------------------------------------------------
__global__ __launch_bounds__(256) void flash_attn_kernel(
    const unsigned short* __restrict__ q, const unsigned short* __restrict__ k,
    const unsigned short* __restrict__ vt, const float* __restrict__ mask,
    unsigned short* __restrict__ ctx) {
  constexpr int KST = HD + 8;  // [key][hd] stride 72 elems = 144 B
  constexpr int VST = 64 + 8;  // [hd][key] stride
  constexpr int PST = 64 + 8;  // [q][key] stride
  __shared__ alignas(16) unsigned short Ksm[2][64 * KST];
  __shared__ alignas(16) unsigned short Vsm[2][64 * VST];
  __shared__ unsigned short Psm[8][16 * PST];

  const int tid = threadIdx.x;
  const int lane = tid & 31, col16 = lane & 15, half = lane >> 4;
  const int w = tid >> 5;
  const int bh = blockIdx.y;
  const int b = bh >> 4, h = bh & 15;
  const int q0 = blockIdx.x * 128;

  // Q fragments for this wave's 16 query rows
  const int qrow = q0 + w * 16 + col16;
  const unsigned short* qp = q + ((size_t)(b * SS + qrow)) * DD + h * HD;
  v16bf qf0 = load_fragA(qp);
  v16bf qf1 = load_fragA(qp + 32);

  FragC o[4];
#pragma unroll
  for (int t = 0; t < 4; ++t)
#pragma unroll
    for (int i = 0; i < 8; ++i) o[t].f[i] = 0.f;
  float m_run[8], l_run[8];
#pragma unroll
  for (int i = 0; i < 8; ++i) { m_run[i] = -3.0e38f; l_run[i] = 0.f; }

  // issue one 64-key K tile + V tile: 4 async b128 ops per thread
  auto issueKV = [&](int buf, int kb) {
#pragma unroll
    for (int it = 0; it < 2; ++it) {
      int c = tid + it * 256;            // 512 chunks of 8 bf16
      int r = c >> 3, c8 = (c & 7) * 8;  // 8 chunks per 64-elem row
      async_ld16(&Ksm[buf][r * KST + c8],
                 k + ((size_t)(b * SS + kb + r)) * DD + h * HD + c8);
      async_ld16(&Vsm[buf][r * VST + c8],
                 vt + (((size_t)(b * HH + h)) * HD + r) * SS + kb + c8);
    }
  };

  issueKV(0, 0);
  for (int kbi = 0; kbi < SS / 64; ++kbi) {
    const int cur = kbi & 1;
    const int kb = kbi * 64;
    if (kbi + 1 < SS / 64) { issueKV(cur ^ 1, kb + 64); wait_async<4>(); }
    else                   { wait_async<0>(); }
    __syncthreads();   // K/V tile [cur] ready

    // scores: S = Q(16x64) * K^T -> 4 tiles of 16x16
    FragC sc[4];
#pragma unroll
    for (int nt = 0; nt < 4; ++nt) {
      v8f sv = {};
      sv = wmma32(qf0, load_fragB(&Ksm[cur][(nt * 16) * KST + 0], KST), sv);
      sv = wmma32(qf1, load_fragB(&Ksm[cur][(nt * 16) * KST + 32], KST), sv);
      sc[nt].v = sv;
      const float mv = mask[b * SS + kb + nt * 16 + col16];
#pragma unroll
      for (int i = 0; i < 8; ++i) sc[nt].f[i] = sc[nt].f[i] * 0.125f + mv;
    }

    // online softmax in accumulator layout (rows i+8*half per lane)
    float rm[8];
#pragma unroll
    for (int i = 0; i < 8; ++i)
      rm[i] = fmaxf(fmaxf(sc[0].f[i], sc[1].f[i]), fmaxf(sc[2].f[i], sc[3].f[i]));
#pragma unroll
    for (int mset = 8; mset; mset >>= 1)
#pragma unroll
      for (int i = 0; i < 8; ++i) rm[i] = fmaxf(rm[i], __shfl_xor(rm[i], mset));

    float alpha[8], rsum[8];
#pragma unroll
    for (int i = 0; i < 8; ++i) {
      float mnew = fmaxf(m_run[i], rm[i]);
      alpha[i] = __expf(m_run[i] - mnew);
      m_run[i] = mnew;
      rsum[i] = 0.f;
    }
#pragma unroll
    for (int nt = 0; nt < 4; ++nt)
#pragma unroll
      for (int i = 0; i < 8; ++i) {
        float p = __expf(sc[nt].f[i] - m_run[i]);
        rsum[i] += p;
        Psm[w][(i + 8 * half) * PST + nt * 16 + col16] = f2bfu(p);
      }
#pragma unroll
    for (int mset = 8; mset; mset >>= 1)
#pragma unroll
      for (int i = 0; i < 8; ++i) rsum[i] += __shfl_xor(rsum[i], mset);
#pragma unroll
    for (int i = 0; i < 8; ++i) l_run[i] = l_run[i] * alpha[i] + rsum[i];

    __syncthreads();   // Psm visible across the wave's lanes

    // O = O*alpha + P(16x64) * V(64keys x 64hd)
    v16bf pa0 = load_fragA(&Psm[w][col16 * PST + 0]);
    v16bf pa1 = load_fragA(&Psm[w][col16 * PST + 32]);
#pragma unroll
    for (int t = 0; t < 4; ++t) {
#pragma unroll
      for (int i = 0; i < 8; ++i) o[t].f[i] *= alpha[i];
      v8f ov = o[t].v;
      ov = wmma32(pa0, load_fragB(&Vsm[cur][(t * 16) * VST + 0], VST), ov);
      ov = wmma32(pa1, load_fragB(&Vsm[cur][(t * 16) * VST + 32], VST), ov);
      o[t].v = ov;
    }
    __syncthreads();   // reads of tile [cur] done before it is re-issued
  }

  // finalize: divide by l, store bf16 ctx token-major
  float linv[8];
#pragma unroll
  for (int i = 0; i < 8; ++i) linv[i] = 1.0f / l_run[i];
  const int qbase = q0 + w * 16;
#pragma unroll
  for (int t = 0; t < 4; ++t)
#pragma unroll
    for (int i = 0; i < 8; ++i) {
      const int qq = qbase + i + 8 * half;
      const int hd = t * 16 + col16;
      ctx[((size_t)(b * SS + qq)) * DD + h * HD + hd] = f2bfu(o[t].f[i] * linv[i]);
    }
}

// ---------------------------------------------------------------------------
// Host orchestration
// ---------------------------------------------------------------------------
extern "C" void kernel_launch(void* const* d_in, const int* in_sizes, int n_in,
                              void* d_out, int out_size, void* d_ws, size_t ws_size,
                              hipStream_t stream) {
  (void)in_sizes; (void)n_in; (void)out_size; (void)ws_size;
  const float* x    = (const float*)d_in[0];
  const float* mask = (const float*)d_in[1];
  const float* wq = (const float*)d_in[2];  const float* bq = (const float*)d_in[3];
  const float* wk = (const float*)d_in[4];  const float* bk = (const float*)d_in[5];
  const float* wv = (const float*)d_in[6];  const float* bv = (const float*)d_in[7];
  const float* wo = (const float*)d_in[8];  const float* bo = (const float*)d_in[9];
  const float* w1 = (const float*)d_in[10]; const float* b1 = (const float*)d_in[11];
  const float* w2 = (const float*)d_in[12]; const float* b2 = (const float*)d_in[13];
  const float* g1 = (const float*)d_in[14]; const float* be1 = (const float*)d_in[15];
  const float* g2 = (const float*)d_in[16]; const float* be2 = (const float*)d_in[17];

  char* ws = (char*)d_ws;
  size_t off = 0;
  auto carve = [&](size_t bytes) {
    void* p = ws + off;
    off += (bytes + 255) & ~(size_t)255;
    return p;
  };
  unsigned short* wqb  = (unsigned short*)carve((size_t)DD * DD * 2);   // [N,K]
  unsigned short* wkb  = (unsigned short*)carve((size_t)DD * DD * 2);
  unsigned short* wvb  = (unsigned short*)carve((size_t)DD * DD * 2);
  unsigned short* wob  = (unsigned short*)carve((size_t)DD * DD * 2);
  unsigned short* w1b  = (unsigned short*)carve((size_t)DD * FFD * 2);
  unsigned short* w2b  = (unsigned short*)carve((size_t)FFD * DD * 2);
  unsigned short* ln1b = (unsigned short*)carve((size_t)NTOK * DD * 2);
  unsigned short* qb   = (unsigned short*)carve((size_t)NTOK * DD * 2);
  unsigned short* kbuf = (unsigned short*)carve((size_t)NTOK * DD * 2);
  unsigned short* vtb  = (unsigned short*)carve((size_t)NTOK * DD * 2);  // [b,h,hd,s]
  unsigned short* ctxb = (unsigned short*)carve((size_t)NTOK * DD * 2);
  float*          hres = (float*)carve((size_t)NTOK * DD * 4);
  unsigned short* ln2b = (unsigned short*)carve((size_t)NTOK * DD * 2);
  unsigned short* ff1b = (unsigned short*)carve((size_t)NTOK * FFD * 2);

  // weight prep: f32 [K,N] -> bf16 transposed [N,K]
  convert_transpose_bf16<<<dim3(DD / 32, DD / 32), 256, 0, stream>>>(wq, wqb, DD, DD);
  convert_transpose_bf16<<<dim3(DD / 32, DD / 32), 256, 0, stream>>>(wk, wkb, DD, DD);
  convert_transpose_bf16<<<dim3(DD / 32, DD / 32), 256, 0, stream>>>(wv, wvb, DD, DD);
  convert_transpose_bf16<<<dim3(DD / 32, DD / 32), 256, 0, stream>>>(wo, wob, DD, DD);
  convert_transpose_bf16<<<dim3(FFD / 32, DD / 32), 256, 0, stream>>>(w1, w1b, DD, FFD);
  convert_transpose_bf16<<<dim3(DD / 32, FFD / 32), 256, 0, stream>>>(w2, w2b, FFD, DD);

  // LN1
  layernorm_bf16_kernel<<<NTOK, 256, 0, stream>>>(x, g1, be1, ln1b);

  // Q,K,V projections
  dim3 gProj(DD / 128, NTOK / 128);
  gemm_bf16_kernel<false, false, 1><<<gProj, 256, 0, stream>>>(
      ln1b, wqb, bq, nullptr, qb, NTOK, DD, DD);
  gemm_bf16_kernel<false, false, 1><<<gProj, 256, 0, stream>>>(
      ln1b, wkb, bk, nullptr, kbuf, NTOK, DD, DD);
  gemm_bf16_kernel<false, false, 2><<<gProj, 256, 0, stream>>>(
      ln1b, wvb, bv, nullptr, vtb, NTOK, DD, DD);

  // attention
  flash_attn_kernel<<<dim3(SS / 128, BB * HH), 256, 0, stream>>>(
      qb, kbuf, vtb, mask, ctxb);

  // output projection + residual (f32)
  gemm_bf16_kernel<false, true, 0><<<gProj, 256, 0, stream>>>(
      ctxb, wob, bo, x, hres, NTOK, DD, DD);

  // LN2
  layernorm_bf16_kernel<<<NTOK, 256, 0, stream>>>(hres, g2, be2, ln2b);

  // FFN1 with exact-erf GELU (bf16 out)
  gemm_bf16_kernel<true, false, 1><<<dim3(FFD / 128, NTOK / 128), 256, 0, stream>>>(
      ln2b, w1b, b1, nullptr, ff1b, NTOK, FFD, DD);

  // FFN2 + residual -> final f32 output
  gemm_bf16_kernel<false, true, 0><<<gProj, 256, 0, stream>>>(
      ff1b, w2b, b2, hres, d_out, NTOK, DD, FFD);
}
